// HPLSTMModule_34179349742199
// MI455X (gfx1250) — compile-verified, matrix operand order checked by reading, not run on hardware
//
#include <hip/hip_runtime.h>
#include <hip/hip_bf16.h>
#include <stdint.h>

// Problem dims (match reference)
#define SEQ_   8192
#define D_IN_  2048
#define D_HID_ 2048
#define D_OUT_ 2048
#define D_CAT_ (D_IN_ + D_HID_)

// Scan chunking
#define NCHUNK 64
#define CHLEN  (SEQ_ / NCHUNK)   // 128

typedef __attribute__((ext_vector_type(16))) __bf16 v16bf;
typedef __attribute__((ext_vector_type(8)))  float  v8f;
typedef __attribute__((ext_vector_type(4)))  int    v4i;

// LDS row stride in bf16 elements: 72 elems = 144B.
// 144B is 16B-aligned (ds_load_b128 legal) and 36-dword stride makes the
// 16 fragment rows hit 16 distinct 4-bank groups -> conflict-free.
#define LDSTR 72

// ---- gfx1250 async global->LDS path (probe-guarded; falls back cleanly) ----
#if defined(__has_builtin)
#if __has_builtin(__builtin_amdgcn_global_load_async_to_lds_b128)
#define HAVE_ASYNC_LDS 1
#endif
#endif

#ifdef HAVE_ASYNC_LDS
typedef __attribute__((address_space(1))) v4i gv4i;   // global int4
typedef __attribute__((address_space(3))) v4i lv4i;   // LDS int4
#endif

__device__ __forceinline__ unsigned short f2bf(float f) {
  unsigned int u = __float_as_uint(f);
  u += 0x7FFFu + ((u >> 16) & 1u);       // round-to-nearest-even
  return (unsigned short)(u >> 16);
}

__device__ __forceinline__ float sigmoidf_(float x) { return 1.0f / (1.0f + __expf(-x)); }
__device__ __forceinline__ float tanh_fast(float x) {
  float e = __expf(2.0f * x);
  return 1.0f - 2.0f / (e + 1.0f);       // correct limits at +/-inf
}

__device__ __forceinline__ v8f vzero8() {
  v8f v;
#pragma unroll
  for (int i = 0; i < 8; ++i) v[i] = 0.0f;
  return v;
}

union FragU { v16bf v; uint4 q[2]; };

// A/B fragment load from an LDS tile stored row-major [row][k0..31], stride LDSTR.
// 16-bit operand layout (ISA 7.12.2): lane l holds row (l&15); elements 0..7 are
// K = 8*(l>>4)+0..7, elements 8..15 are K = 16 + 8*(l>>4)+0..7.
__device__ __forceinline__ v16bf frag_ld(const unsigned short* tile, int row, int lane) {
  const unsigned short* p = tile + row * LDSTR + ((lane >> 4) << 3);
  FragU u;
  u.q[0] = *(const uint4*)(p);
  u.q[1] = *(const uint4*)(p + 16);
  return u.v;
}

__device__ __forceinline__ v8f wmma_bf16(v16bf a, v16bf b, v8f c) {
  // (neg_a, A, neg_b, B, c_mod, C, reuse_a, reuse_b)
  return __builtin_amdgcn_wmma_f32_16x16x32_bf16(false, a, false, b, (short)0, c, false, false);
}

// Cooperative copy of a rows x 32 bf16 tile, global (row-major, ld elems) -> LDS.
// Async path: GLOBAL_LOAD_ASYNC_TO_LDS_B128 (ASYNCcnt, no VGPR staging).
__device__ __forceinline__ void tile_copy(unsigned short* dst, const unsigned short* src,
                                          int ld, int rows, int tid, int nthreads) {
  int nchunks = rows * 4;                       // four 16B chunks per row
  for (int c = tid; c < nchunks; c += nthreads) {
    int row = c >> 2;
    int kq  = (c & 3) << 3;                     // element offset within row
#ifdef HAVE_ASYNC_LDS
    __builtin_amdgcn_global_load_async_to_lds_b128(
        (gv4i*)(src + (size_t)row * ld + kq),
        (lv4i*)(dst + row * LDSTR + kq),
        /*offset=*/0, /*cpol=*/0);
#else
    uint4 v = *(const uint4*)(src + (size_t)row * ld + kq);
    *(uint4*)(dst + row * LDSTR + kq) = v;
#endif
  }
}

// Wait for this wave's outstanding async tile copies (block barrier follows at call site).
__device__ __forceinline__ void tile_copy_fence() {
#ifdef HAVE_ASYNC_LDS
#if __has_builtin(__builtin_amdgcn_s_wait_asynccnt)
  __builtin_amdgcn_s_wait_asynccnt(0);
#else
  asm volatile("s_wait_asynccnt 0" ::: "memory");
#endif
#endif
}

// ---------------------------------------------------------------- converts --

__global__ void cvt_f32_bf16_kernel(const float* __restrict__ src,
                                    unsigned short* __restrict__ dst, int n) {
  int i4 = (blockIdx.x * blockDim.x + threadIdx.x) * 4;
  if (i4 + 3 < n) {
    float4 v = *(const float4*)(src + i4);
    dst[i4 + 0] = f2bf(v.x);
    dst[i4 + 1] = f2bf(v.y);
    dst[i4 + 2] = f2bf(v.z);
    dst[i4 + 3] = f2bf(v.w);
  }
}

// W (K x N, f32 row-major) -> WT (N x K, bf16 row-major)
__global__ __launch_bounds__(256) void transpose_cvt_kernel(
    const float* __restrict__ W, unsigned short* __restrict__ WT, int K, int N) {
  __shared__ float t[32][33];
  int tx = threadIdx.x & 31;
  int ty = threadIdx.x >> 5;                 // 0..7
  int x0 = blockIdx.x * 32;                  // N dimension
  int y0 = blockIdx.y * 32;                  // K dimension
#pragma unroll
  for (int i = 0; i < 4; ++i) {
    int y = y0 + ty + i * 8;
    t[ty + i * 8][tx] = W[(size_t)y * N + (x0 + tx)];
  }
  __syncthreads();
#pragma unroll
  for (int i = 0; i < 4; ++i) {
    int r = ty + i * 8;
    WT[(size_t)(x0 + r) * K + (y0 + tx)] = f2bf(t[tx][r]);
  }
}

// ------------------------------------------------------------- gates GEMM --
// fg = sigmoid(x@Wf + bf), hr = sigmoid(x@Wi + bi) * tanh(x@Wh + bh)
// Block tile 128x64, 8 waves (4x2), wave tile 32x32 (2x2 WMMA tiles), 3 B sets.
// Double-buffered LDS: async loads of tile k+1 overlap WMMA on tile k.

__global__ __launch_bounds__(256) void gates_gemm_kernel(
    const unsigned short* __restrict__ xb,
    const unsigned short* __restrict__ WfT,
    const unsigned short* __restrict__ WiT,
    const unsigned short* __restrict__ WhT,
    const float* __restrict__ bf_, const float* __restrict__ bi_,
    const float* __restrict__ bh_,
    float* __restrict__ fg, float* __restrict__ hr) {
  __shared__ unsigned short sA [2][128 * LDSTR];
  __shared__ unsigned short sBf[2][ 64 * LDSTR];
  __shared__ unsigned short sBi[2][ 64 * LDSTR];
  __shared__ unsigned short sBh[2][ 64 * LDSTR];

  const int tid  = threadIdx.x;
  const int lane = tid & 31;
  const int wave = tid >> 5;
  const int wm   = wave >> 1;                // 0..3
  const int wn   = wave & 1;                 // 0..1
  const int m0   = blockIdx.y * 128;
  const int n0   = blockIdx.x * 64;

  const unsigned short* xrow  = xb  + (size_t)m0 * D_IN_;
  const unsigned short* wfrow = WfT + (size_t)n0 * D_IN_;
  const unsigned short* wirow = WiT + (size_t)n0 * D_IN_;
  const unsigned short* whrow = WhT + (size_t)n0 * D_IN_;

  v8f acc[3][2][2];
#pragma unroll
  for (int g = 0; g < 3; ++g)
#pragma unroll
    for (int tm = 0; tm < 2; ++tm)
#pragma unroll
      for (int tn = 0; tn < 2; ++tn) acc[g][tm][tn] = vzero8();

  auto issue_tiles = [&](int buf, int k0) {
    tile_copy(sA[buf],  xrow  + k0, D_IN_, 128, tid, 256);
    tile_copy(sBf[buf], wfrow + k0, D_IN_,  64, tid, 256);
    tile_copy(sBi[buf], wirow + k0, D_IN_,  64, tid, 256);
    tile_copy(sBh[buf], whrow + k0, D_IN_,  64, tid, 256);
  };

  issue_tiles(0, 0);
  tile_copy_fence();
  __syncthreads();

  int it = 0;
  for (int k0 = 0; k0 < D_IN_; k0 += 32, ++it) {
    const int cur = it & 1;
    if (k0 + 32 < D_IN_) issue_tiles(cur ^ 1, k0 + 32);   // overlap with WMMA below
    if (k0 + 64 < D_IN_)
      __builtin_prefetch(xrow + (k0 + 64) + (size_t)tid * 8, 0, 1);

    v16bf a[2], bF[2], bI[2], bH[2];
#pragma unroll
    for (int tm = 0; tm < 2; ++tm) a[tm] = frag_ld(sA[cur], wm * 32 + tm * 16, lane);
#pragma unroll
    for (int tn = 0; tn < 2; ++tn) {
      bF[tn] = frag_ld(sBf[cur], wn * 32 + tn * 16, lane);
      bI[tn] = frag_ld(sBi[cur], wn * 32 + tn * 16, lane);
      bH[tn] = frag_ld(sBh[cur], wn * 32 + tn * 16, lane);
    }
#pragma unroll
    for (int tm = 0; tm < 2; ++tm)
#pragma unroll
      for (int tn = 0; tn < 2; ++tn) {
        acc[0][tm][tn] = wmma_bf16(a[tm], bF[tn], acc[0][tm][tn]);
        acc[1][tm][tn] = wmma_bf16(a[tm], bI[tn], acc[1][tm][tn]);
        acc[2][tm][tn] = wmma_bf16(a[tm], bH[tn], acc[2][tm][tn]);
      }

    tile_copy_fence();          // next tile landed in LDS
    __syncthreads();            // all waves done reading cur / writing nxt
  }

  // Epilogue. C/D layout: lane (l&15)=N col, VGPR r -> M = r + 8*(l>>4).
#pragma unroll
  for (int tm = 0; tm < 2; ++tm)
#pragma unroll
    for (int tn = 0; tn < 2; ++tn) {
      int n  = n0 + wn * 32 + tn * 16 + (lane & 15);
      int mb = m0 + wm * 32 + tm * 16 + ((lane >> 4) << 3);
      float vbf = bf_[n], vbi = bi_[n], vbh = bh_[n];
#pragma unroll
      for (int r = 0; r < 8; ++r) {
        size_t idx = (size_t)(mb + r) * D_HID_ + n;
        float uf = acc[0][tm][tn][r] + vbf;
        float ui = acc[1][tm][tn][r] + vbi;
        float uh = acc[2][tm][tn][r] + vbh;
        fg[idx] = sigmoidf_(uf);
        hr[idx] = sigmoidf_(ui) * tanh_fast(uh);
      }
    }
}

// ------------------------------------------------------------------- scan --
// c_t = fg_t * c_{t-1} + hr_t  as chunked scan. Pass1 in-place: fg->cumF, hr->local c.

__global__ void scan_pass1_kernel(float* fgFl, float* hrCl,
                                  float* chunkF, float* chunkC) {
  int h  = blockIdx.x * blockDim.x + threadIdx.x;   // hidden channel
  int ch = blockIdx.y;
  int s0 = ch * CHLEN;
  float F = 1.0f, c = 0.0f;
  for (int i = 0; i < CHLEN; ++i) {
    size_t idx = (size_t)(s0 + i) * D_HID_ + h;
    float f = fgFl[idx];
    float v = hrCl[idx];
    c = fmaf(f, c, v);
    F = F * f;
    fgFl[idx] = F;       // cumulative forget product within chunk
    hrCl[idx] = c;       // local inclusive scan
  }
  chunkF[(size_t)ch * D_HID_ + h] = F;
  chunkC[(size_t)ch * D_HID_ + h] = c;
}

__global__ void scan_pass2_kernel(const float* __restrict__ chunkF,
                                  const float* __restrict__ chunkC,
                                  float* __restrict__ carry) {
  int h = blockIdx.x * blockDim.x + threadIdx.x;
  float c = 0.0f;
  for (int k = 0; k < NCHUNK; ++k) {
    size_t idx = (size_t)k * D_HID_ + h;
    carry[idx] = c;                       // carry entering chunk k
    c = fmaf(chunkF[idx], c, chunkC[idx]);
  }
}

// c = c_local + cumF * carry ; write fp32 in place and bf16 copy for the GEMM.
__global__ void scan_pass3_kernel(const float* __restrict__ Fl, float* clC,
                                  const float* __restrict__ carry,
                                  unsigned short* __restrict__ cb) {
  size_t i = (size_t)blockIdx.x * blockDim.x + threadIdx.x;
  int h  = (int)(i & (D_HID_ - 1));
  int s  = (int)(i >> 11);                // D_HID_ = 2^11
  int ch = s >> 7;                        // CHLEN = 128
  float c = fmaf(Fl[i], carry[(size_t)ch * D_HID_ + h], clC[i]);
  clC[i] = c;
  cb[i]  = f2bf(c);
}

// ---------------------------------------------------------------- og GEMM --
// og = sigmoid([x;c] @ Wo + bo); g = og * c   (g stored bf16 for final GEMM)

__global__ __launch_bounds__(256) void og_gemm_kernel(
    const unsigned short* __restrict__ xb,
    const unsigned short* __restrict__ cb,
    const unsigned short* __restrict__ WoT,   // (D_HID x D_CAT) bf16 row-major
    const float* __restrict__ bo_,
    const float* __restrict__ cfp,            // c fp32
    unsigned short* __restrict__ gb) {
  __shared__ unsigned short sA[2][128 * LDSTR];
  __shared__ unsigned short sB[2][ 64 * LDSTR];

  const int tid  = threadIdx.x;
  const int lane = tid & 31;
  const int wave = tid >> 5;
  const int wm   = wave >> 1;
  const int wn   = wave & 1;
  const int m0   = blockIdx.y * 128;
  const int n0   = blockIdx.x * 64;

  v8f acc[2][2];
#pragma unroll
  for (int tm = 0; tm < 2; ++tm)
#pragma unroll
    for (int tn = 0; tn < 2; ++tn) acc[tm][tn] = vzero8();

  auto issue_tiles = [&](int buf, int k0) {
    const unsigned short* Asrc =
        (k0 < D_IN_) ? (xb + (size_t)m0 * D_IN_ + k0)
                     : (cb + (size_t)m0 * D_HID_ + (k0 - D_IN_));
    tile_copy(sA[buf], Asrc, (k0 < D_IN_) ? D_IN_ : D_HID_, 128, tid, 256);
    tile_copy(sB[buf], WoT + (size_t)n0 * D_CAT_ + k0, D_CAT_, 64, tid, 256);
  };

  issue_tiles(0, 0);
  tile_copy_fence();
  __syncthreads();

  int it = 0;
  for (int k0 = 0; k0 < D_CAT_; k0 += 32, ++it) {
    const int cur = it & 1;
    if (k0 + 32 < D_CAT_) issue_tiles(cur ^ 1, k0 + 32);

    v16bf a[2], b[2];
#pragma unroll
    for (int tm = 0; tm < 2; ++tm) a[tm] = frag_ld(sA[cur], wm * 32 + tm * 16, lane);
#pragma unroll
    for (int tn = 0; tn < 2; ++tn) b[tn] = frag_ld(sB[cur], wn * 32 + tn * 16, lane);
#pragma unroll
    for (int tm = 0; tm < 2; ++tm)
#pragma unroll
      for (int tn = 0; tn < 2; ++tn)
        acc[tm][tn] = wmma_bf16(a[tm], b[tn], acc[tm][tn]);

    tile_copy_fence();
    __syncthreads();
  }

#pragma unroll
  for (int tm = 0; tm < 2; ++tm)
#pragma unroll
    for (int tn = 0; tn < 2; ++tn) {
      int n  = n0 + wn * 32 + tn * 16 + (lane & 15);
      int mb = m0 + wm * 32 + tm * 16 + ((lane >> 4) << 3);
      float vb = bo_[n];
#pragma unroll
      for (int r = 0; r < 8; ++r) {
        size_t idx = (size_t)(mb + r) * D_HID_ + n;
        float og = sigmoidf_(acc[tm][tn][r] + vb);
        gb[idx] = f2bf(og * cfp[idx]);
      }
    }
}

// --------------------------------------------------------------- out GEMM --
// o = g @ Wout + bout (fp32 output)

__global__ __launch_bounds__(256) void out_gemm_kernel(
    const unsigned short* __restrict__ gb,
    const unsigned short* __restrict__ WoutT,  // (D_OUT x D_HID) bf16 row-major
    const float* __restrict__ bout_,
    float* __restrict__ out) {
  __shared__ unsigned short sA[2][128 * LDSTR];
  __shared__ unsigned short sB[2][ 64 * LDSTR];

  const int tid  = threadIdx.x;
  const int lane = tid & 31;
  const int wave = tid >> 5;
  const int wm   = wave >> 1;
  const int wn   = wave & 1;
  const int m0   = blockIdx.y * 128;
  const int n0   = blockIdx.x * 64;

  v8f acc[2][2];
#pragma unroll
  for (int tm = 0; tm < 2; ++tm)
#pragma unroll
    for (int tn = 0; tn < 2; ++tn) acc[tm][tn] = vzero8();

  auto issue_tiles = [&](int buf, int k0) {
    tile_copy(sA[buf], gb    + (size_t)m0 * D_HID_ + k0, D_HID_, 128, tid, 256);
    tile_copy(sB[buf], WoutT + (size_t)n0 * D_HID_ + k0, D_HID_,  64, tid, 256);
  };

  issue_tiles(0, 0);
  tile_copy_fence();
  __syncthreads();

  int it = 0;
  for (int k0 = 0; k0 < D_HID_; k0 += 32, ++it) {
    const int cur = it & 1;
    if (k0 + 32 < D_HID_) issue_tiles(cur ^ 1, k0 + 32);

    v16bf a[2], b[2];
#pragma unroll
    for (int tm = 0; tm < 2; ++tm) a[tm] = frag_ld(sA[cur], wm * 32 + tm * 16, lane);
#pragma unroll
    for (int tn = 0; tn < 2; ++tn) b[tn] = frag_ld(sB[cur], wn * 32 + tn * 16, lane);
#pragma unroll
    for (int tm = 0; tm < 2; ++tm)
#pragma unroll
      for (int tn = 0; tn < 2; ++tn)
        acc[tm][tn] = wmma_bf16(a[tm], b[tn], acc[tm][tn]);

    tile_copy_fence();
    __syncthreads();
  }

#pragma unroll
  for (int tm = 0; tm < 2; ++tm)
#pragma unroll
    for (int tn = 0; tn < 2; ++tn) {
      int n  = n0 + wn * 32 + tn * 16 + (lane & 15);
      int mb = m0 + wm * 32 + tm * 16 + ((lane >> 4) << 3);
      float vb = bout_[n];
#pragma unroll
      for (int r = 0; r < 8; ++r)
        out[(size_t)(mb + r) * D_OUT_ + n] = acc[tm][tn][r] + vb;
    }
}

// ------------------------------------------------------------------ driver --

extern "C" void kernel_launch(void* const* d_in, const int* in_sizes, int n_in,
                              void* d_out, int out_size, void* d_ws, size_t ws_size,
                              hipStream_t stream) {
  (void)in_sizes; (void)n_in; (void)out_size; (void)ws_size;

  const float* x     = (const float*)d_in[0];
  const float* Wf    = (const float*)d_in[1];
  const float* bf_   = (const float*)d_in[2];
  const float* Wi    = (const float*)d_in[3];
  const float* bi_   = (const float*)d_in[4];
  const float* Wh    = (const float*)d_in[5];
  const float* bh_   = (const float*)d_in[6];
  const float* Wo    = (const float*)d_in[7];
  const float* bo_   = (const float*)d_in[8];
  const float* Wout  = (const float*)d_in[9];
  const float* bout_ = (const float*)d_in[10];
  float* out = (float*)d_out;

  // Workspace layout (~280 MB total)
  char*  ws  = (char*)d_ws;
  size_t off = 0;
  auto alloc = [&](size_t bytes) -> void* {
    void* p = ws + off;
    off += (bytes + 255) & ~(size_t)255;
    return p;
  };
  unsigned short* xb    = (unsigned short*)alloc((size_t)SEQ_  * D_IN_  * 2);
  unsigned short* WfT   = (unsigned short*)alloc((size_t)D_HID_ * D_IN_ * 2);
  unsigned short* WiT   = (unsigned short*)alloc((size_t)D_HID_ * D_IN_ * 2);
  unsigned short* WhT   = (unsigned short*)alloc((size_t)D_HID_ * D_IN_ * 2);
  unsigned short* WoT   = (unsigned short*)alloc((size_t)D_HID_ * D_CAT_ * 2);
  unsigned short* WoutT = (unsigned short*)alloc((size_t)D_OUT_ * D_HID_ * 2);
  float* bufA   = (float*)alloc((size_t)SEQ_ * D_HID_ * 4);   // fg -> cumF
  float* bufB   = (float*)alloc((size_t)SEQ_ * D_HID_ * 4);   // hr -> local c -> c
  unsigned short* cb = (unsigned short*)alloc((size_t)SEQ_ * D_HID_ * 2);
  unsigned short* gb = (unsigned short*)alloc((size_t)SEQ_ * D_HID_ * 2);
  float* chunkF = (float*)alloc((size_t)NCHUNK * D_HID_ * 4);
  float* chunkC = (float*)alloc((size_t)NCHUNK * D_HID_ * 4);
  float* carry  = (float*)alloc((size_t)NCHUNK * D_HID_ * 4);

  // 1) Precision conversion (one pass; weights stay resident in 192MB L2 after)
  {
    int n = SEQ_ * D_IN_;
    cvt_f32_bf16_kernel<<<n / (256 * 4), 256, 0, stream>>>(x, xb, n);
  }
  transpose_cvt_kernel<<<dim3(D_HID_ / 32, D_IN_ / 32), 256, 0, stream>>>(Wf, WfT, D_IN_, D_HID_);
  transpose_cvt_kernel<<<dim3(D_HID_ / 32, D_IN_ / 32), 256, 0, stream>>>(Wi, WiT, D_IN_, D_HID_);
  transpose_cvt_kernel<<<dim3(D_HID_ / 32, D_IN_ / 32), 256, 0, stream>>>(Wh, WhT, D_IN_, D_HID_);
  transpose_cvt_kernel<<<dim3(D_HID_ / 32, D_CAT_ / 32), 256, 0, stream>>>(Wo, WoT, D_CAT_, D_HID_);
  transpose_cvt_kernel<<<dim3(D_OUT_ / 32, D_HID_ / 32), 256, 0, stream>>>(Wout, WoutT, D_HID_, D_OUT_);

  // 2) Fused gate GEMMs (x loaded once for three weights)
  gates_gemm_kernel<<<dim3(D_HID_ / 64, SEQ_ / 128), 256, 0, stream>>>(
      xb, WfT, WiT, WhT, bf_, bi_, bh_, bufA, bufB);

  // 3) Chunked linear-recurrence scan
  scan_pass1_kernel<<<dim3(D_HID_ / 256, NCHUNK), 256, 0, stream>>>(bufA, bufB, chunkF, chunkC);
  scan_pass2_kernel<<<D_HID_ / 256, 256, 0, stream>>>(chunkF, chunkC, carry);
  scan_pass3_kernel<<<(SEQ_ * D_HID_) / 256, 256, 0, stream>>>(bufA, bufB, carry, cb);

  // 4) Output-gate GEMM over concat [x; c], fused og*c
  og_gemm_kernel<<<dim3(D_HID_ / 64, SEQ_ / 128), 256, 0, stream>>>(
      xb, cb, WoT, bo_, bufB, gb);

  // 5) Final projection
  out_gemm_kernel<<<dim3(D_OUT_ / 64, SEQ_ / 128), 256, 0, stream>>>(
      gb, WoutT, bout_, out);
}
